// Block_33354716020840
// MI455X (gfx1250) — matrix operand order repaired and checked by reading, not compiled, not run on hardware
//
#include <hip/hip_runtime.h>

// ---------------------------------------------------------------------------
// Types for CDNA5 WMMA (wave32): v_wmma_f32_16x16x32_bf16
// ---------------------------------------------------------------------------
typedef __attribute__((ext_vector_type(16))) __bf16 v16bf;
typedef __attribute__((ext_vector_type(8)))  __bf16 v8bf;
typedef __attribute__((ext_vector_type(8)))  float  v8f;
typedef int b128i __attribute__((vector_size(16)));   // matches async-LDS builtin param

union FragBf { v16bf v; v8bf h[2]; };

__device__ __forceinline__ v8f wmma_bf16(const FragBf& a, const FragBf& b, v8f c) {
    return __builtin_amdgcn_wmma_f32_16x16x32_bf16(
        /*neg_a=*/false, a.v, /*neg_b=*/false, b.v,
        /*c_mod=*/(short)0, c, /*reuse_a=*/false, /*reuse_b=*/false);
}

__device__ __forceinline__ v8f vzero8() {
    v8f z;
#pragma unroll
    for (int i = 0; i < 8; ++i) z[i] = 0.f;
    return z;
}

// ---------------------------------------------------------------------------
// Async global->LDS (ASYNCcnt path). Guarded: falls back to plain copies.
// Builtin signature (from probe): (v4i* global, v4i* lds, Imm offset, Imm cpol)
// ---------------------------------------------------------------------------
#if defined(__has_builtin)
#  if __has_builtin(__builtin_amdgcn_global_load_async_to_lds_b128) && \
      __has_builtin(__builtin_amdgcn_s_wait_asynccnt)
#    define USE_ASYNC_LDS 1
#  endif
#endif
#ifndef USE_ASYNC_LDS
#  define USE_ASYNC_LDS 0
#endif

__device__ __forceinline__ void copy16_g2l(__bf16* ldst, const __bf16* gsrc) {
#if USE_ASYNC_LDS
    __builtin_amdgcn_global_load_async_to_lds_b128(
        (b128i*)gsrc, (b128i*)ldst, /*offset=*/0, /*cpol=*/0);
#else
    *(v8bf*)ldst = *(const v8bf*)gsrc;
#endif
}

template <int N> __device__ __forceinline__ void async_wait_cnt() {
#if USE_ASYNC_LDS
    __builtin_amdgcn_s_wait_asynccnt(N);
#endif
}

// ---------------------------------------------------------------------------
// Problem constants
// ---------------------------------------------------------------------------
static constexpr int Bb  = 4;
static constexpr int Tt  = 2048;
static constexpr int Cc  = 2048;
static constexpr int Dd  = 128;     // C / H
static constexpr int FFf = 8192;    // 4C
static constexpr int BT  = Bb * Tt; // 8192 rows
static constexpr int Hh  = 16;

// ---------------------------------------------------------------------------
// fp32 [K][N] -> bf16 [N][K]   (weights: transpose so B-operand is contiguous-K)
// ---------------------------------------------------------------------------
__global__ void cast_transpose(const float* __restrict__ src, __bf16* __restrict__ dst,
                               int K, int N) {
    long long idx = (long long)blockIdx.x * 256 + threadIdx.x;
    long long tot = (long long)K * N;
    if (idx >= tot) return;
    int k = (int)(idx / N);
    int n = (int)(idx % N);
    dst[(size_t)n * K + k] = (__bf16)src[idx];
}

// ---------------------------------------------------------------------------
// RMSNorm + cast to bf16. One block (256 thr = 8 waves) per row of C=2048.
// ---------------------------------------------------------------------------
__global__ __launch_bounds__(256)
void rmsnorm_cast(const float* __restrict__ x, const float* __restrict__ g,
                  __bf16* __restrict__ o, int C) {
    __shared__ float red[8];
    const int row = blockIdx.x;
    const float* xr = x + (size_t)row * C;
    float ss = 0.f;
    for (int c = threadIdx.x; c < C; c += 256) { float v = xr[c]; ss += v * v; }
#pragma unroll
    for (int m = 16; m >= 1; m >>= 1) ss += __shfl_xor(ss, m, 32);
    if ((threadIdx.x & 31) == 0) red[threadIdx.x >> 5] = ss;
    __syncthreads();
    float tot = 0.f;
#pragma unroll
    for (int i = 0; i < 8; ++i) tot += red[i];
    float rs = rsqrtf(tot / (float)C + 1e-6f);
    __bf16* orow = o + (size_t)row * C;
    for (int c = threadIdx.x; c < C; c += 256) orow[c] = (__bf16)(xr[c] * rs * g[c]);
}

// ---------------------------------------------------------------------------
// SwiGLU elementwise: h = silu(gate) * val   (may alias gate in-place)
// ---------------------------------------------------------------------------
__global__ void swiglu_kernel(const __bf16* __restrict__ g, const __bf16* __restrict__ v,
                              __bf16* __restrict__ h, long long n) {
    long long i = (long long)blockIdx.x * 256 + threadIdx.x;
    if (i >= n) return;
    float gf = (float)g[i], vf = (float)v[i];
    h[i] = (__bf16)((gf / (1.f + __expf(-gf))) * vf);
}

// ---------------------------------------------------------------------------
// Tiled bf16 GEMM: out[M][N] = A[M][K](bf16) x Bt[N][K](bf16, pre-transposed)
//   EPI==0: store bf16 ; EPI==1: store fp32 = acc + res[M][N]
// Block tile 128x256, BK=32, 8 waves (2 in M x 4 in N), wave tile 64x64.
// Double-buffered LDS staged via async global->LDS copies (6 per thread/tile);
// s_wait_asynccnt<=6 covers exactly the previously-issued tile (in-order ctr).
// LDS row stride 40 elems (80 B) -> 16B-aligned b128, conflict-free.
// ---------------------------------------------------------------------------
static constexpr int LDT = 40;

__device__ __forceinline__ void gemm_issue_tile(
    const __bf16* __restrict__ A, const __bf16* __restrict__ Bt, int K,
    int m0, int n0, int k0, __bf16* as, __bf16* bs, int tid) {
#pragma unroll
    for (int it = 0; it < 2; ++it) {                    // A: 128x32 = 512 chunks
        int c = tid + it * 256;
        int r = c >> 2, col = (c & 3) * 8;
        copy16_g2l(&as[r * LDT + col], A + (size_t)(m0 + r) * K + k0 + col);
    }
#pragma unroll
    for (int it = 0; it < 4; ++it) {                    // B: 256x32 = 1024 chunks
        int c = tid + it * 256;
        int r = c >> 2, col = (c & 3) * 8;
        copy16_g2l(&bs[r * LDT + col], Bt + (size_t)(n0 + r) * K + k0 + col);
    }
}

template <int EPI>
__global__ __launch_bounds__(256)
void gemm_bf16(const __bf16* __restrict__ A, const __bf16* __restrict__ Bt,
               int M, int N, int K, const float* __restrict__ res,
               void* __restrict__ outv) {
    __shared__ __bf16 As[2][128 * LDT];
    __shared__ __bf16 Bs[2][256 * LDT];

    const int tid  = threadIdx.x;
    const int lane = tid & 31, wave = tid >> 5;
    const int half = lane >> 4, l16 = lane & 15;
    const int wm = wave >> 2, wn = wave & 3;
    const int m0 = blockIdx.y * 128, n0 = blockIdx.x * 256;

    v8f acc[4][4];
#pragma unroll
    for (int i = 0; i < 4; ++i)
#pragma unroll
        for (int j = 0; j < 4; ++j) acc[i][j] = vzero8();

    const int KT = K / 32;
    gemm_issue_tile(A, Bt, K, m0, n0, 0, As[0], Bs[0], tid);

    for (int kt = 0; kt < KT; ++kt) {
        const int cur = kt & 1;
        if (kt + 1 < KT) {
            gemm_issue_tile(A, Bt, K, m0, n0, (kt + 1) * 32,
                            As[1 - cur], Bs[1 - cur], tid);
            async_wait_cnt<6>();   // only the 6 just-issued may remain
        } else {
            async_wait_cnt<0>();
        }
        __syncthreads();           // tile kt visible to all waves

        const __bf16* as = As[cur];
        const __bf16* bs = Bs[cur];

        FragBf fa[4];
#pragma unroll
        for (int i = 0; i < 4; ++i) {
            int m = wm * 64 + i * 16 + l16;
            fa[i].h[0] = *(const v8bf*)&as[m * LDT + half * 8];       // K 0..7 / 8..15
            fa[i].h[1] = *(const v8bf*)&as[m * LDT + 16 + half * 8];  // K 16..23 / 24..31
        }
#pragma unroll
        for (int j = 0; j < 4; ++j) {
            int n = wn * 64 + j * 16 + l16;
            FragBf fb;
            fb.h[0] = *(const v8bf*)&bs[n * LDT + half * 8];          // K 0..7  / 8..15
            fb.h[1] = *(const v8bf*)&bs[n * LDT + 16 + half * 8];     // K 16..23 / 24..31
#pragma unroll
            for (int i = 0; i < 4; ++i)
                acc[i][j] = wmma_bf16(fa[i], fb, acc[i][j]);
        }
        __syncthreads();           // reads done before buf is overwritten
    }

    // Epilogue: C/D layout -> element (M = v + 8*half, N = l16)
#pragma unroll
    for (int i = 0; i < 4; ++i)
#pragma unroll
        for (int j = 0; j < 4; ++j)
#pragma unroll
            for (int v = 0; v < 8; ++v) {
                int gm = m0 + wm * 64 + i * 16 + v + 8 * half;
                int gn = n0 + wn * 64 + j * 16 + l16;
                float val = acc[i][j][v];
                if constexpr (EPI == 0) {
                    ((__bf16*)outv)[(size_t)gm * N + gn] = (__bf16)val;
                } else {
                    ((float*)outv)[(size_t)gm * N + gn] =
                        val + res[(size_t)gm * N + gn];
                }
            }
}

// ---------------------------------------------------------------------------
// Causal flash attention. qkv bf16 [B,T,3C] (q|k|v, head h at cols h*D).
// Block = (b,h, q-tile of 128). 8 waves, each owns 16 q rows.
// Q fragments live in registers; K tile + V^T tile in LDS; P via per-wave LDS.
// ---------------------------------------------------------------------------
__global__ __launch_bounds__(256)
void attention_kernel(const __bf16* __restrict__ qkv, __bf16* __restrict__ y) {
    constexpr int LK = 136;  // K-tile row stride (d) -> 272 B, 16B aligned
    constexpr int LV = 72;   // V^T row stride (kv)  -> 144 B
    constexpr int LP = 72;   // P row stride
    __shared__ __bf16 Ks[64 * LK];      // [kv][d]
    __shared__ __bf16 Vt[128 * LV];     // [d][kv]
    __shared__ __bf16 Ps[8 * 16 * LP];  // per-wave [16][kv]

    const int tid = threadIdx.x, lane = tid & 31, w = tid >> 5;
    const int half = lane >> 4, l16 = lane & 15;
    const int bh = blockIdx.x, b = bh >> 4, h = bh & 15;
    const int qt = blockIdx.y, q0 = qt * 128;
    const size_t rowstride = 3 * Cc;
    const size_t base = (size_t)b * Tt * rowstride;

    // Q fragments (16 rows x 128 d = 4 k-steps), direct from global
    FragBf qa[4];
    {
        const size_t qrow = base + (size_t)(q0 + w * 16 + l16) * rowstride + h * Dd;
#pragma unroll
        for (int d = 0; d < 4; ++d) {
            qa[d].h[0] = *(const v8bf*)(qkv + qrow + d * 32 + half * 8);
            qa[d].h[1] = *(const v8bf*)(qkv + qrow + d * 32 + 16 + half * 8);
        }
    }

    v8f c_o[8];
#pragma unroll
    for (int f = 0; f < 8; ++f) c_o[f] = vzero8();
    float mrow[8], lrow[8];
#pragma unroll
    for (int v = 0; v < 8; ++v) { mrow[v] = -1e30f; lrow[v] = 0.f; }

    const int nkv = (qt + 1) * 2;  // kv tiles of 64 covering [0, q0+128)
    for (int kvt = 0; kvt < nkv; ++kvt) {
        const int kv0 = kvt * 64;
        __syncthreads();
        // load K tile (64x128) and transposed V tile: 1024 chunks of 8 bf16
#pragma unroll
        for (int it = 0; it < 4; ++it) {
            int c = tid + it * 256;
            int r = c >> 4, dcol = (c & 15) * 8;
            size_t gk = base + (size_t)(kv0 + r) * rowstride + Cc + h * Dd + dcol;
            *(v8bf*)&Ks[r * LK + dcol] = *(const v8bf*)(qkv + gk);
            v8bf vv = *(const v8bf*)(qkv + gk + Cc);
#pragma unroll
            for (int j = 0; j < 8; ++j) Vt[(dcol + j) * LV + r] = vv[j];
        }
        __syncthreads();

        // S = Q K^T : 4 kv-subtiles x 4 d-steps = 16 WMMA
        v8f s[4];
#pragma unroll
        for (int a = 0; a < 4; ++a) s[a] = vzero8();
#pragma unroll
        for (int a = 0; a < 4; ++a) {
            int n = a * 16 + l16;
#pragma unroll
            for (int d = 0; d < 4; ++d) {
                FragBf kb;
                kb.h[0] = *(const v8bf*)&Ks[n * LK + d * 32 + half * 8];
                kb.h[1] = *(const v8bf*)&Ks[n * LK + d * 32 + 16 + half * 8];
                s[a] = wmma_bf16(qa[d], kb, s[a]);
            }
        }

        // scale + causal mask (element: q = q0+w*16+v+8*half, kv = kv0+a*16+l16)
        const float sc = 0.0883883476483184f;  // 1/sqrt(128)
#pragma unroll
        for (int a = 0; a < 4; ++a)
#pragma unroll
            for (int v = 0; v < 8; ++v) {
                int q  = q0 + w * 16 + v + 8 * half;
                int kv = kv0 + a * 16 + l16;
                float sv = s[a][v] * sc;
                s[a][v] = (kv > q) ? -1e30f : sv;
            }

        // online softmax: rows live across 16 lanes (width-16 reductions)
#pragma unroll
        for (int v = 0; v < 8; ++v) {
            float mx = fmaxf(fmaxf(s[0][v], s[1][v]), fmaxf(s[2][v], s[3][v]));
#pragma unroll
            for (int off = 8; off >= 1; off >>= 1)
                mx = fmaxf(mx, __shfl_xor(mx, off, 16));
            float mnew = fmaxf(mrow[v], mx);
            float al = __expf(mrow[v] - mnew);
            float rs = 0.f;
#pragma unroll
            for (int a = 0; a < 4; ++a) {
                float p = __expf(s[a][v] - mnew);
                s[a][v] = p;
                rs += p;
            }
#pragma unroll
            for (int off = 8; off >= 1; off >>= 1) rs += __shfl_xor(rs, off, 16);
            lrow[v] = lrow[v] * al + rs;
            mrow[v] = mnew;
#pragma unroll
            for (int f = 0; f < 8; ++f) c_o[f][v] *= al;
        }

        // P (16x64) -> per-wave LDS in A-fragment feedable layout
#pragma unroll
        for (int a = 0; a < 4; ++a)
#pragma unroll
            for (int v = 0; v < 8; ++v)
                Ps[(w * 16 + v + 8 * half) * LP + a * 16 + l16] = (__bf16)s[a][v];

        // O += P x V : 2 kv-steps x 8 d-frags = 16 WMMA
#pragma unroll
        for (int kks = 0; kks < 2; ++kks) {
            int kk = kks * 32;
            FragBf pa;
            pa.h[0] = *(const v8bf*)&Ps[(w * 16 + l16) * LP + kk + half * 8];
            pa.h[1] = *(const v8bf*)&Ps[(w * 16 + l16) * LP + kk + 16 + half * 8];
#pragma unroll
            for (int f = 0; f < 8; ++f) {
                FragBf vb;
                vb.h[0] = *(const v8bf*)&Vt[(f * 16 + l16) * LV + kk + half * 8];
                vb.h[1] = *(const v8bf*)&Vt[(f * 16 + l16) * LV + kk + 16 + half * 8];
                c_o[f] = wmma_bf16(pa, vb, c_o[f]);
            }
        }
    }

    // normalize + store bf16 y [B,T,C] at col h*D + d
#pragma unroll
    for (int f = 0; f < 8; ++f)
#pragma unroll
        for (int v = 0; v < 8; ++v) {
            int q = q0 + w * 16 + v + 8 * half;
            int d = f * 16 + l16;
            float o = c_o[f][v] / lrow[v];
            y[(size_t)(b * Tt + q) * Cc + h * Dd + d] = (__bf16)o;
        }
}

// ---------------------------------------------------------------------------
// Orchestration
// ---------------------------------------------------------------------------
extern "C" void kernel_launch(void* const* d_in, const int* in_sizes, int n_in,
                              void* d_out, int out_size, void* d_ws, size_t ws_size,
                              hipStream_t stream) {
    (void)in_sizes; (void)n_in; (void)out_size; (void)ws_size;

    const float* x      = (const float*)d_in[0];
    const float* g1     = (const float*)d_in[1];
    const float* g2     = (const float*)d_in[2];
    const float* w_qkv  = (const float*)d_in[3];
    const float* w_proj = (const float*)d_in[4];
    const float* w_gate = (const float*)d_in[5];
    const float* w_val  = (const float*)d_in[6];
    const float* w_out  = (const float*)d_in[7];
    float* out = (float*)d_out;

    char* ws = (char*)d_ws;
    size_t off = 0;
    auto carve = [&](size_t bytes) -> char* {
        char* p = ws + off;
        off += (bytes + 255) & ~(size_t)255;
        return p;
    };

    __bf16* wqkv_t  = (__bf16*)carve((size_t)3 * Cc * Cc * 2);  // [3C][C]
    __bf16* wproj_t = (__bf16*)carve((size_t)Cc * Cc * 2);      // [C][C]
    __bf16* wgate_t = (__bf16*)carve((size_t)FFf * Cc * 2);     // [FF][C]
    __bf16* wval_t  = (__bf16*)carve((size_t)FFf * Cc * 2);     // [FF][C]
    __bf16* wout_t  = (__bf16*)carve((size_t)Cc * FFf * 2);     // [C][FF]
    __bf16* xn      = (__bf16*)carve((size_t)BT * Cc * 2);      // rmsnorm out (reused)
    char*   r1      = carve((size_t)BT * FFf * 2);              // qkv -> gate -> h
    char*   r2      = carve((size_t)BT * FFf * 2);              // y_attn -> val
    float*  x1      = (float*)carve((size_t)BT * Cc * 4);       // residual after attn

    __bf16* qkv   = (__bf16*)r1;
    __bf16* gate  = (__bf16*)r1;
    __bf16* hbuf  = (__bf16*)r1;
    __bf16* yattn = (__bf16*)r2;
    __bf16* valb  = (__bf16*)r2;

    auto launch_t = [&](const float* src, __bf16* dst, int K, int N) {
        long long tot = (long long)K * N;
        cast_transpose<<<(unsigned)((tot + 255) / 256), 256, 0, stream>>>(src, dst, K, N);
    };
    launch_t(w_qkv,  wqkv_t,  Cc,  3 * Cc);
    launch_t(w_proj, wproj_t, Cc,  Cc);
    launch_t(w_gate, wgate_t, Cc,  FFf);
    launch_t(w_val,  wval_t,  Cc,  FFf);
    launch_t(w_out,  wout_t,  FFf, Cc);

    // x1 = x + attn(rmsnorm(x,g1))
    rmsnorm_cast<<<BT, 256, 0, stream>>>(x, g1, xn, Cc);
    gemm_bf16<0><<<dim3(3 * Cc / 256, BT / 128), 256, 0, stream>>>(
        xn, wqkv_t, BT, 3 * Cc, Cc, nullptr, qkv);
    attention_kernel<<<dim3(Bb * Hh, Tt / 128), 256, 0, stream>>>(qkv, yattn);
    gemm_bf16<1><<<dim3(Cc / 256, BT / 128), 256, 0, stream>>>(
        yattn, wproj_t, BT, Cc, Cc, x, x1);

    // out = x1 + swiglu(rmsnorm(x1,g2))
    rmsnorm_cast<<<BT, 256, 0, stream>>>(x1, g2, xn, Cc);
    gemm_bf16<0><<<dim3(FFf / 256, BT / 128), 256, 0, stream>>>(
        xn, wgate_t, BT, FFf, Cc, nullptr, gate);
    gemm_bf16<0><<<dim3(FFf / 256, BT / 128), 256, 0, stream>>>(
        xn, wval_t, BT, FFf, Cc, nullptr, valb);
    {
        long long tot = (long long)BT * FFf;
        swiglu_kernel<<<(unsigned)((tot + 255) / 256), 256, 0, stream>>>(gate, valb, hbuf, tot);
    }
    gemm_bf16<1><<<dim3(Cc / 256, BT / 128), 256, 0, stream>>>(
        hbuf, wout_t, BT, Cc, FFf, x1, out);
}
